// FusedModConv_88330297410400
// MI455X (gfx1250) — matrix operand order repaired and checked by reading, not compiled, not run on hardware
//
#include <hip/hip_runtime.h>
#include <hip/hip_bf16.h>

// ---------------------------------------------------------------------------
// StyleGAN2 modulated 3x3 conv, B=16, CIN=COUT=512, H=W=64, WDIM=512.
// Implicit-GEMM conv on v_wmma_f32_16x16x32_bf16 with a 4-row x 66-col LDS
// halo per workgroup (A staged once per cin-chunk; taps = LDS offset shifts)
// and double-buffered B tiles. Wave subtile 32x64 = 2x4 WMMA accumulators
// (1.5 LDS b128 loads per WMMA).
// ---------------------------------------------------------------------------

#define BATCH 16
#define CIN   512
#define COUT  512
#define HW    4096      // 64*64
#define SDIM  512       // style vector dim
#define NTAP  9

typedef __attribute__((ext_vector_type(16))) __bf16 v16bf;
typedef __attribute__((ext_vector_type(8)))  __bf16 v8bf;
typedef __attribute__((ext_vector_type(8)))  float  v8f;

#define COEF_MOD  0.04419417382415922f   // 1/sqrt(512)
#define COEF_CONV 0.014731391274719739f  // 1/sqrt(9*512)

// LDS geometry (element units, bf16)
#define ACOL 40                 // per-pixel cin stride (32 used + 8 pad)
#define AROW (66 * ACOL)        // halo row stride: 66 cols
#define BROW 40                 // per-cout cin stride (32 used + 8 pad)
#define BSIZE (128 * BROW)      // one B buffer (128 couts x 32 cin)

// ---------------------------------------------------------------------------
// Kernel 1: style[b][cin] = coef_mod * (w[b,:] . mod_w[:,cin]) + mod_b[cin] + 1
// ---------------------------------------------------------------------------
__global__ __launch_bounds__(256) void style_kernel(
    const float* __restrict__ w, const float* __restrict__ mod_w,
    const float* __restrict__ mod_b, float* __restrict__ style) {
  int idx = blockIdx.x * 256 + threadIdx.x;      // b*512 + cin
  int b   = idx >> 9;
  int cin = idx & 511;
  const float* wr = w + b * SDIM;
  float s = 0.f;
  for (int d = 0; d < SDIM; ++d) s += wr[d] * mod_w[d * CIN + cin];
  style[idx] = s * COEF_MOD + mod_b[cin] + 1.0f;
}

// ---------------------------------------------------------------------------
// Kernel 2: dvec[b][cout] = rsqrt(coef^2 * sum_{tap,cin}(conv_w*style)^2 + 1e-8)
// ---------------------------------------------------------------------------
__global__ __launch_bounds__(256) void demod_kernel(
    const float* __restrict__ conv_w, const float* __restrict__ style,
    float* __restrict__ dvec) {
  int idx  = blockIdx.x * 256 + threadIdx.x;     // b*512 + cout
  int b    = idx >> 9;
  int cout = idx & 511;
  const float* st = style + b * CIN;
  float s = 0.f;
  for (int kk = 0; kk < NTAP; ++kk) {
    const float* cw = conv_w + (size_t)kk * CIN * COUT + cout;
    for (int cin = 0; cin < CIN; ++cin) {
      float v = cw[(size_t)cin * COUT] * st[cin];
      s += v * v;
    }
  }
  s *= COEF_CONV * COEF_CONV;
  dvec[idx] = rsqrtf(s + 1e-8f);
}

// ---------------------------------------------------------------------------
// Kernel 3: wT[b][tap][cout][cin] = bf16(coef*conv_w[tap,cin,cout]
//                                        * style[b,cin] * dvec[b,cout])
// 32x32 LDS-tiled transpose; both global sides coalesced.
// ---------------------------------------------------------------------------
__global__ __launch_bounds__(256) void wmod_kernel(
    const float* __restrict__ conv_w, const float* __restrict__ style,
    const float* __restrict__ dvec, __bf16* __restrict__ wT) {
  __shared__ float tile[32][33];
  int bz = blockIdx.z;
  int b = bz / NTAP, kk = bz % NTAP;
  int coutBase = blockIdx.x * 32;
  int cinBase  = blockIdx.y * 32;
  int tx = threadIdx.x;          // 0..31
  int ty = threadIdx.y;          // 0..7
#pragma unroll
  for (int r = 0; r < 4; ++r) {
    int cin  = cinBase + ty + r * 8;
    int cout = coutBase + tx;
    float v = conv_w[((size_t)kk * CIN + cin) * COUT + cout] *
              style[b * CIN + cin] * COEF_CONV;
    tile[ty + r * 8][tx] = v;
  }
  __syncthreads();
#pragma unroll
  for (int r = 0; r < 4; ++r) {
    int cout = coutBase + ty + r * 8;
    int cin  = cinBase + tx;
    float v = tile[tx][ty + r * 8] * dvec[b * COUT + cout];
    wT[(((size_t)(b * NTAP + kk) * COUT + cout) * CIN) + cin] = (__bf16)v;
  }
}

// ---------------------------------------------------------------------------
// Kernel 4: implicit-GEMM conv per batch with halo-staged A.
//   WG tile: 128 pixels (2 image rows) x 128 couts, 8 waves; each wave a
//   32x64 subtile = 2x4 WMMA f32_16x16x32_bf16 accumulators.
// ---------------------------------------------------------------------------
__global__ __launch_bounds__(256) void conv_kernel(
    const float* __restrict__ x, const __bf16* __restrict__ wT,
    float* __restrict__ y) {
  __shared__ __bf16 Ah[4 * AROW];        // 21120 B
  __shared__ __bf16 Bs[2 * BSIZE];       // 20480 B

  const int b  = blockIdx.z;
  const int m0 = blockIdx.x * 128;       // pixel tile base (2 rows)
  const int h0 = m0 >> 6;                // first image row of tile
  const int n0 = blockIdx.y * 128;       // cout tile base
  const int t    = threadIdx.x;
  const int lane = t & 31;
  const int wv   = t >> 5;
  const int wM   = wv & 3;               // 4 waves along M (32 each)
  const int wN   = wv >> 2;              // 2 waves along N (64 each)

  v8f acc[2][4] = {};

  // halo loader: one core pixel per thread (4 rows x 64 cols)
  const int hr = t >> 6;                 // 0..3 halo row
  const int hc = t & 63;                 // 0..63 image col
  const int hh = h0 - 1 + hr;
  const bool rowok = (unsigned)hh < 64u;

  // B loader: thread -> cout = t>>1 (0..127), two 8-bf16 chunks at (t&1)*16
  const int bn = t >> 1;
  const int bk = (t & 1) * 16;

  const float*  xb = x  + (size_t)b * CIN * HW;
  const __bf16* wb = wT + (size_t)b * NTAP * COUT * CIN + (size_t)n0 * CIN;

  // ---- zero the always-outside halo border cols (ww = -1 and ww = 64), once
  {
    int zr = t >> 6;                     // 0..3
    int zs = (t >> 5) & 1;               // side
    int zc = t & 31;                     // cin slot
    Ah[zr * AROW + (zs ? 65 * ACOL : 0) + zc] = (__bf16)0.f;
  }

  // per-lane fragment base offsets (element units)
  int baseA[2], baseB[4];
  {
    const int kbA = (lane < 16) ? 0 : 16;  // A: lanes>=16 hold K 16..31
    const int kbB = (lane < 16) ? 0 : 16;  // B: lanes>=16 hold K 16..31
#pragma unroll
    for (int tM = 0; tM < 2; ++tM) {
      int m  = wM * 32 + tM * 16 + (lane & 15);   // 0..127
      int lr = m >> 6;                             // tile-local row 0/1
      int c  = m & 63;
      baseA[tM] = lr * AROW + (c + 1) * ACOL + kbA;  // taps add row/col shift
    }
#pragma unroll
    for (int tN = 0; tN < 4; ++tN)
      baseB[tN] = (wN * 64 + tN * 16 + (lane & 15)) * BROW + kbB;
  }

  for (int c0 = 0; c0 < CIN; c0 += 32) {
    __syncthreads();   // previous chunk's Ah/Bs reads complete

    // ---- stage A halo core: 32 coalesced b32 loads, 4x ds_store_b128
    {
      float vals[32];
      if (rowok) {
        const float* xp = xb + (size_t)c0 * HW + (hh << 6) + hc;
#pragma unroll
        for (int i = 0; i < 32; ++i) vals[i] = xp[(size_t)i * HW];
      } else {
#pragma unroll
        for (int i = 0; i < 32; ++i) vals[i] = 0.f;
      }
      __bf16* dst = &Ah[hr * AROW + (hc + 1) * ACOL];
#pragma unroll
      for (int i = 0; i < 4; ++i) {
        v8bf v;
#pragma unroll
        for (int j = 0; j < 8; ++j) v[j] = (__bf16)vals[i * 8 + j];
        *(v8bf*)(dst + i * 8) = v;
      }
    }
    // ---- stage B tile for tap 0 into buffer 0 (2x b128 per thread)
    {
      const __bf16* src = wb + (size_t)bn * CIN + (c0 + bk);
      *(v8bf*)&Bs[bn * BROW + bk]     = *(const v8bf*)src;
      *(v8bf*)&Bs[bn * BROW + bk + 8] = *(const v8bf*)(src + 8);
    }

    for (int kk = 0; kk < NTAP; ++kk) {
      __syncthreads();

      // issue next tap's B-tile global loads early (double buffer)
      v8bf breg0, breg1;
      const bool havenext = (kk < NTAP - 1);
      if (havenext) {
        const __bf16* src =
            wb + (size_t)(kk + 1) * COUT * CIN + (size_t)bn * CIN + (c0 + bk);
        breg0 = *(const v8bf*)src;
        breg1 = *(const v8bf*)(src + 8);
      }

      // ---- fragments: tap = pure LDS offset shift into the halo
      const int offA = (kk / 3) * AROW + (kk % 3) * ACOL;
      const int offB = (kk & 1) * BSIZE;
      v16bf afrag[2], bfrag[4];
#pragma unroll
      for (int tM = 0; tM < 2; ++tM) {
        const __bf16* ap = &Ah[baseA[tM] + offA];
        v8bf lo = *(const v8bf*)(ap);
        v8bf hi = *(const v8bf*)(ap + 8);
        v16bf f;
#pragma unroll
        for (int i = 0; i < 8; ++i) { f[i] = lo[i]; f[i + 8] = hi[i]; }
        afrag[tM] = f;
      }
#pragma unroll
      for (int tN = 0; tN < 4; ++tN) {
        const __bf16* bp = &Bs[offB + baseB[tN]];
        v8bf lo = *(const v8bf*)(bp);
        v8bf hi = *(const v8bf*)(bp + 8);
        v16bf f;
#pragma unroll
        for (int i = 0; i < 8; ++i) { f[i] = lo[i]; f[i + 8] = hi[i]; }
        bfrag[tN] = f;
      }

#pragma unroll
      for (int tM = 0; tM < 2; ++tM)
#pragma unroll
        for (int tN = 0; tN < 4; ++tN)
          acc[tM][tN] = __builtin_amdgcn_wmma_f32_16x16x32_bf16(
              false, afrag[tM], false, bfrag[tN],
              (short)0, acc[tM][tN], false, false);

      // park next tap's B tile into the other buffer
      if (havenext) {
        __bf16* d2 = &Bs[((kk + 1) & 1) * BSIZE + bn * BROW + bk];
        *(v8bf*)(d2)     = breg0;
        *(v8bf*)(d2 + 8) = breg1;
      }
    }
  }

  // ---- epilogue: each lane stores 8 consecutive pixels at fixed cout
  float* yb = y + (size_t)b * COUT * HW;
#pragma unroll
  for (int tM = 0; tM < 2; ++tM) {
    int mBase = m0 + wM * 32 + tM * 16 + ((lane >= 16) ? 8 : 0);
#pragma unroll
    for (int tN = 0; tN < 4; ++tN) {
      int cout = n0 + wN * 64 + tN * 16 + (lane & 15);
      float* dst = yb + (size_t)cout * HW + mBase;
      v8f c = acc[tM][tN];
      *(float4*)(dst)     = make_float4(c[0], c[1], c[2], c[3]);
      *(float4*)(dst + 4) = make_float4(c[4], c[5], c[6], c[7]);
    }
  }
}

// ---------------------------------------------------------------------------
// Launch: style -> demod -> modulated-weight transform -> WMMA conv.
// Workspace: style 32KB @0, dvec 32KB @32K, wT (bf16) 75.5MB @64K.
// ---------------------------------------------------------------------------
extern "C" void kernel_launch(void* const* d_in, const int* in_sizes, int n_in,
                              void* d_out, int out_size, void* d_ws, size_t ws_size,
                              hipStream_t stream) {
  const float* x      = (const float*)d_in[0];
  const float* w      = (const float*)d_in[1];
  const float* conv_w = (const float*)d_in[2];
  const float* mod_w  = (const float*)d_in[3];
  const float* mod_b  = (const float*)d_in[4];
  float* y = (float*)d_out;

  char* ws = (char*)d_ws;
  float*  style = (float*)(ws);                 // 16*512 f32
  float*  dvec  = (float*)(ws + 32768);         // 16*512 f32
  __bf16* wT    = (__bf16*)(ws + 65536);        // 16*9*512*512 bf16

  style_kernel<<<(BATCH * CIN) / 256, 256, 0, stream>>>(w, mod_w, mod_b, style);
  demod_kernel<<<(BATCH * COUT) / 256, 256, 0, stream>>>(conv_w, style, dvec);
  wmod_kernel<<<dim3(COUT / 32, CIN / 32, BATCH * NTAP), dim3(32, 8), 0, stream>>>(
      conv_w, style, dvec, wT);
  conv_kernel<<<dim3(HW / 128, COUT / 128, BATCH), 256, 0, stream>>>(x, wT, y);
}
